// CausalSelfAttention_80126909874598
// MI455X (gfx1250) — matrix operand order repaired and checked by reading, not compiled, not run on hardware
//
#include <hip/hip_runtime.h>

// ---------------- problem constants (match reference) ----------------
static constexpr int Bc  = 2;
static constexpr int Tc  = 2048;
static constexpr int NEc = 1024;
static constexpr int NHc = 16;
static constexpr int NKVc = 4;
static constexpr int HDc = 64;
static constexpr int GCc = 32;
static constexpr int WINc = 1024;
static constexpr int Mrows = Bc * Tc;          // 4096

#ifndef __has_builtin
#define __has_builtin(x) 0
#endif
#if __has_builtin(__builtin_amdgcn_global_load_async_to_lds_b128)
#define HAVE_ASYNC_LDS 1
#else
#define HAVE_ASYNC_LDS 0
#endif
#define AS1 __attribute__((address_space(1)))
#define AS3 __attribute__((address_space(3)))

// ---------------- WMMA / async types ----------------
typedef __bf16 v16bf __attribute__((ext_vector_type(16)));
typedef float  v8f   __attribute__((ext_vector_type(8)));
typedef int    v4i   __attribute__((ext_vector_type(4)));

union V16 {
    v16bf v;
    unsigned short u[16];
    unsigned int   w[8];
};

__device__ inline v8f wmma_bf16(const V16& a, const V16& b, v8f c) {
    return __builtin_amdgcn_wmma_f32_16x16x32_bf16(
        /*neg_a=*/false, a.v, /*neg_b=*/false, b.v,
        /*c_mod=*/(short)0, c, /*reuse_a=*/false, /*reuse_b=*/false);
}

// A-matrix 16x32 bf16 K-mapping (ISA 7.12.2): VGPR p, lane-half h
__device__ inline int kmapA(int p, int half) {
    return (p < 4) ? (half * 8 + 2 * p) : (16 + half * 8 + 2 * (p - 4));
}
// B-matrix 32x16 bf16 K-mapping (same K order as A)
__device__ inline int kmapB(int p, int half) {
    return (p < 4) ? (half * 8 + 2 * p) : (16 + half * 8 + 2 * (p - 4));
}

// f32 -> bf16 (round-to-nearest-even)
__device__ inline unsigned short f2b(float f) {
    unsigned u = __builtin_bit_cast(unsigned, f);
    unsigned r = u + 0x7FFFu + ((u >> 16) & 1u);
    return (unsigned short)(r >> 16);
}

// ---------------- kernel 0: f32 -> bf16 convert ----------------
__global__ void cvt_f32_bf16(const float* __restrict__ s,
                             unsigned short* __restrict__ d, int n) {
    int i = blockIdx.x * blockDim.x + threadIdx.x;
    int stride = gridDim.x * blockDim.x;
    for (; i < n; i += stride) d[i] = f2b(s[i]);
}

// ---------------- bf16 WMMA GEMM, C = A(MxK) * W(KxN), f32 out -------------
// block: 256 threads (8 waves); 64x64 tile; double-buffered LDS; A tile is
// staged with GLOBAL_LOAD_ASYNC_TO_LDS_B128 when available (ASYNCcnt path).
static constexpr int SAs = 40;   // A LDS row stride (ushorts): 80B, 16B-aligned
static constexpr int SBs = 34;   // B^T LDS row stride (ushorts)

__global__ __launch_bounds__(256)
void gemm_bf16(const unsigned short* __restrict__ A,
               const unsigned short* __restrict__ W,
               float* __restrict__ C, int M, int N, int K) {
    __shared__ __align__(16) unsigned short As[2][64 * SAs];   // [m][k]
    __shared__ __align__(16) unsigned short Bs[2][64 * SBs];   // [n][k]

    const int tid  = threadIdx.x;
    const int lane = tid & 31, wave = tid >> 5;
    const int half = lane >> 4, l15 = lane & 15;
    const int mblk = blockIdx.y * 64, nblk = blockIdx.x * 64;
    const int tile_m  = wave & 3;            // 0..3 (16-row tile)
    const int tile_n0 = (wave >> 2) * 2;     // 0 or 2 (two 16-col tiles)

    const int arow = tid >> 2;               // A-stage: 64 rows x (4 thr/row)
    const int akc  = (tid & 3) * 8;          // 8 bf16 = 16B per thread
    const int brow = tid >> 3;               // W-stage: 32 k-rows x (8 thr/row)
    const int bnc  = (tid & 7) * 8;

    auto stage = [&](int kb, int buf) {
        // ---- A tile 64x32 ----
        const unsigned short* asrc = A + (size_t)(mblk + arow) * K + kb + akc;
        unsigned short* adst = &As[buf][arow * SAs + akc];
#if HAVE_ASYNC_LDS
        __builtin_amdgcn_global_load_async_to_lds_b128(
            (AS1 v4i*)asrc, (AS3 v4i*)adst, 0, 0);
#else
        {
            const unsigned int* s32 = (const unsigned int*)asrc;
            unsigned int* d32 = (unsigned int*)adst;
            d32[0] = s32[0]; d32[1] = s32[1]; d32[2] = s32[2]; d32[3] = s32[3];
        }
#endif
        // ---- W tile 32x64, transposed into Bs[n][k] ----
        const unsigned short* wsrc = W + (size_t)(kb + brow) * N + nblk + bnc;
#pragma unroll
        for (int e = 0; e < 8; ++e) Bs[buf][(bnc + e) * SBs + brow] = wsrc[e];
    };

    v8f acc0 = {}; v8f acc1 = {};
    const int nsteps = K >> 5;
    stage(0, 0);                                   // prologue

    for (int ki = 0; ki < nsteps; ++ki) {
        const int buf = ki & 1;
#if HAVE_ASYNC_LDS
        asm volatile("s_wait_asynccnt 0" ::: "memory");
#endif
        __syncthreads();                            // staged tile visible
        if (ki + 1 < nsteps) stage((ki + 1) << 5, buf ^ 1);

        V16 af;
#pragma unroll
        for (int p = 0; p < 8; ++p)
            af.w[p] = *(const unsigned int*)
                &As[buf][(tile_m * 16 + l15) * SAs + kmapA(p, half)];
        V16 bf0, bf1;
#pragma unroll
        for (int p = 0; p < 8; ++p) {
            int kk = kmapB(p, half);
            bf0.w[p] = *(const unsigned int*)&Bs[buf][(tile_n0 * 16 + l15) * SBs + kk];
            bf1.w[p] = *(const unsigned int*)&Bs[buf][((tile_n0 + 1) * 16 + l15) * SBs + kk];
        }
        acc0 = wmma_bf16(af, bf0, acc0);
        acc1 = wmma_bf16(af, bf1, acc1);
    }

#pragma unroll
    for (int g = 0; g < 8; ++g) {
        int row  = mblk + tile_m * 16 + g + 8 * half;
        int col0 = nblk + tile_n0 * 16 + l15;
        C[(size_t)row * N + col0]      = acc0[g];
        C[(size_t)row * N + col0 + 16] = acc1[g];
    }
}

// ---------------- kernel 2: rope + rmsnorm + gated-ve, f32 -> bf16 ----------
__global__ __launch_bounds__(128)
void rope_norm_gate(const float* __restrict__ x, const float* __restrict__ ve,
                    const float* __restrict__ cosb, const float* __restrict__ sinb,
                    const float* __restrict__ Wg,
                    const float* __restrict__ qf, const float* __restrict__ kf,
                    const float* __restrict__ vf,
                    unsigned short* __restrict__ qb, unsigned short* __restrict__ kbn,
                    unsigned short* __restrict__ vbn) {
    const int bt = blockIdx.x;                      // b*T + t
    const int tid = threadIdx.x;
    const int wave = tid >> 5, lane = tid & 31;
    __shared__ float gate[NKVc];

    const float c = cosb[(size_t)bt * 32 + lane];
    const float s = sinb[(size_t)bt * 32 + lane];

    if (tid < NKVc) {                                // gate = 2*sigmoid(x[:32]@Wg)
        float acc = 0.f;
        for (int g = 0; g < GCc; ++g)
            acc += x[(size_t)bt * NEc + g] * Wg[g * NKVc + tid];
        gate[tid] = 2.f / (1.f + __expf(-acc));
    }
    __syncthreads();

    for (int h = wave; h < NHc; h += 4) {            // Q heads
        size_t base = ((size_t)bt * NHc + h) * HDc;
        float x1 = qf[base + lane], x2 = qf[base + 32 + lane];
        float r1 =  x1 * c + x2 * s;
        float r2 = -x1 * s + x2 * c;
        float ss = r1 * r1 + r2 * r2;
        for (int m = 16; m >= 1; m >>= 1) ss += __shfl_xor(ss, m, 32);
        float inv = rsqrtf(ss * (1.f / 64.f) + 1e-6f);
        qb[base + lane]      = f2b(r1 * inv);
        qb[base + 32 + lane] = f2b(r2 * inv);
    }
    if (wave < NKVc) {                               // K heads
        int h = wave;
        size_t base = ((size_t)bt * NKVc + h) * HDc;
        float x1 = kf[base + lane], x2 = kf[base + 32 + lane];
        float r1 =  x1 * c + x2 * s;
        float r2 = -x1 * s + x2 * c;
        float ss = r1 * r1 + r2 * r2;
        for (int m = 16; m >= 1; m >>= 1) ss += __shfl_xor(ss, m, 32);
        float inv = rsqrtf(ss * (1.f / 64.f) + 1e-6f);
        kbn[base + lane]      = f2b(r1 * inv);
        kbn[base + 32 + lane] = f2b(r2 * inv);
    }
    for (int idx = tid; idx < NKVc * HDc; idx += 128) {   // V + gate*ve
        int kh = idx >> 6;
        size_t off = (size_t)bt * (NKVc * HDc) + idx;
        vbn[off] = f2b(vf[off] + gate[kh] * ve[off]);
    }
}

// ---------------- kernel 3: flash attention (windowed causal, GQA) ----------
// grid (T/64, NH, B); block 128 (4 waves); wave owns a 16-row query tile.
__global__ __launch_bounds__(128)
void attn(const unsigned short* __restrict__ qb,
          const unsigned short* __restrict__ kbn,
          const unsigned short* __restrict__ vbn,
          unsigned short* __restrict__ yb) {
    __shared__ __align__(16) unsigned short pb[4][16 * 34];  // per-wave P scratch

    const int wave = threadIdx.x >> 5, lane = threadIdx.x & 31;
    const int half = lane >> 4, l15 = lane & 15;
    const int h = blockIdx.y, b = blockIdx.z;
    const int kh = h >> 2;                       // rep = NH/NKV = 4
    const int qbase = blockIdx.x * 64 + wave * 16;
    unsigned short* mypb = pb[wave];

    // Q fragments (16x64 -> two 16x32 A fragments), straight from global bf16
    V16 qa[2];
    {
        const unsigned short* qrow =
            qb + (((size_t)(b * Tc + qbase + l15) * NHc + h) * HDc);
#pragma unroll
        for (int cchunk = 0; cchunk < 2; ++cchunk)
#pragma unroll
            for (int p = 0; p < 8; ++p)
                qa[cchunk].w[p] =
                    *(const unsigned int*)(qrow + cchunk * 32 + kmapA(p, half));
    }

    float mrow[8], lrow[8];
    v8f o[4] = {v8f{}, v8f{}, v8f{}, v8f{}};
#pragma unroll
    for (int g = 0; g < 8; ++g) { mrow[g] = -1e30f; lrow[g] = 0.f; }

    int j0 = qbase - WINc; if (j0 < 0) j0 = 0; j0 &= ~31;
    const int qlast = qbase + 15;

    for (int jb = j0; jb <= qlast; jb += 32) {
        // prefetch next key/value tile rows (global_prefetch_b8)
        {
            int tn = jb + 32 + l15; if (tn > Tc - 1) tn = Tc - 1;
            __builtin_prefetch(kbn + (((size_t)(b * Tc + tn) * NKVc + kh) * HDc), 0, 3);
            __builtin_prefetch(vbn + (((size_t)(b * Tc + tn) * NKVc + kh) * HDc), 0, 3);
        }
        // ---- S = Q * K^T for 32 keys (two 16-key B tiles) ----
        v8f s0 = {}, s1 = {};
#pragma unroll
        for (int n = 0; n < 2; ++n) {
            int tk = jb + n * 16 + l15; if (tk > Tc - 1) tk = Tc - 1;
            const unsigned short* krow =
                kbn + (((size_t)(b * Tc + tk) * NKVc + kh) * HDc);
            V16 kf0, kf1;
#pragma unroll
            for (int p = 0; p < 8; ++p) {
                int km = kmapB(p, half);
                kf0.w[p] = *(const unsigned int*)(krow + km);
                kf1.w[p] = *(const unsigned int*)(krow + 32 + km);
            }
            if (n == 0) { s0 = wmma_bf16(qa[0], kf0, s0); s0 = wmma_bf16(qa[1], kf1, s0); }
            else        { s1 = wmma_bf16(qa[0], kf0, s1); s1 = wmma_bf16(qa[1], kf1, s1); }
        }

        // ---- mask + online softmax (row reductions across 16-lane halves) --
        float alpha[8];
#pragma unroll
        for (int g = 0; g < 8; ++g) {
            const int i = qbase + g + 8 * half;
            const int jc0 = jb + l15, jc1 = jb + 16 + l15;
            float v0 = s0[g] * 0.125f;
            float v1 = s1[g] * 0.125f;
            v0 = ((jc0 <= i) && (i - jc0 <= WINc)) ? v0 : -1e30f;
            v1 = ((jc1 <= i) && (i - jc1 <= WINc)) ? v1 : -1e30f;
            float rmax = fmaxf(v0, v1);
            for (int xm = 8; xm >= 1; xm >>= 1)
                rmax = fmaxf(rmax, __shfl_xor(rmax, xm, 32));
            float mnew = fmaxf(mrow[g], rmax);
            float a = __expf(mrow[g] - mnew);
            float p0 = __expf(v0 - mnew), p1 = __expf(v1 - mnew);
            float rs = p0 + p1;
            for (int xm = 8; xm >= 1; xm >>= 1) rs += __shfl_xor(rs, xm, 32);
            lrow[g] = lrow[g] * a + rs;
            mrow[g] = mnew;
            alpha[g] = a;
            mypb[(g + 8 * half) * 34 + l15]      = f2b(p0);   // C-layout -> LDS
            mypb[(g + 8 * half) * 34 + 16 + l15] = f2b(p1);
        }
#pragma unroll
        for (int cc = 0; cc < 4; ++cc)
#pragma unroll
            for (int g = 0; g < 8; ++g) o[cc][g] *= alpha[g];

        asm volatile("s_wait_dscnt 0" ::: "memory");

        // ---- re-swizzle P to A-fragment layout, then O += P * V ----
        V16 pa;
#pragma unroll
        for (int p = 0; p < 8; ++p)
            pa.w[p] = *(const unsigned int*)&mypb[l15 * 34 + kmapA(p, half)];

#pragma unroll
        for (int cc = 0; cc < 4; ++cc) {
            V16 vfrag;
#pragma unroll
            for (int p = 0; p < 8; ++p) {
                int km = kmapB(p, half);
                int t0 = jb + km;     if (t0 > Tc - 1) t0 = Tc - 1;
                int t1 = jb + km + 1; if (t1 > Tc - 1) t1 = Tc - 1;
                unsigned lo = vbn[(((size_t)(b * Tc + t0) * NKVc + kh) * HDc) + cc * 16 + l15];
                unsigned hi = vbn[(((size_t)(b * Tc + t1) * NKVc + kh) * HDc) + cc * 16 + l15];
                vfrag.w[p] = lo | (hi << 16);
            }
            o[cc] = wmma_bf16(pa, vfrag, o[cc]);
        }
    }

    // ---- normalize + store y as bf16 ----
#pragma unroll
    for (int g = 0; g < 8; ++g) {
        float invl = 1.f / lrow[g];
        int t = qbase + g + 8 * half;
        size_t base = ((size_t)(b * Tc + t) * NHc + h) * HDc;
#pragma unroll
        for (int cc = 0; cc < 4; ++cc)
            yb[base + cc * 16 + l15] = f2b(o[cc][g] * invl);
    }
}

// ---------------- host-side launch ----------------
extern "C" void kernel_launch(void* const* d_in, const int* in_sizes, int n_in,
                              void* d_out, int out_size, void* d_ws, size_t ws_size,
                              hipStream_t stream) {
    const float* x    = (const float*)d_in[0];
    const float* ve   = (const float*)d_in[1];
    const float* cosb = (const float*)d_in[2];
    const float* sinb = (const float*)d_in[3];
    const float* Wq   = (const float*)d_in[4];
    const float* Wk   = (const float*)d_in[5];
    const float* Wv   = (const float*)d_in[6];
    const float* Wo   = (const float*)d_in[7];
    const float* Wg   = (const float*)d_in[8];
    float* out = (float*)d_out;

    char* ws = (char*)d_ws;
    size_t off = 0;
    auto take = [&](size_t bytes) -> void* {
        void* p = ws + off;
        off += (bytes + 255) & ~(size_t)255;
        return p;
    };

    const int nX  = Mrows * NEc;          // 4,194,304
    const int nWq = NEc * NHc * HDc;      // 1,048,576
    const int nWk = NEc * NKVc * HDc;     //   262,144
    const int nWo = NEc * NEc;            // 1,048,576

    unsigned short* xb  = (unsigned short*)take((size_t)nX  * 2);
    unsigned short* wqb = (unsigned short*)take((size_t)nWq * 2);
    unsigned short* wkb = (unsigned short*)take((size_t)nWk * 2);
    unsigned short* wvb = (unsigned short*)take((size_t)nWk * 2);
    unsigned short* wob = (unsigned short*)take((size_t)nWo * 2);
    float* qf = (float*)take((size_t)Mrows * NHc  * HDc * 4);
    float* kf = (float*)take((size_t)Mrows * NKVc * HDc * 4);
    float* vf = (float*)take((size_t)Mrows * NKVc * HDc * 4);
    unsigned short* qbb = (unsigned short*)take((size_t)Mrows * NHc  * HDc * 2);
    unsigned short* kbb = (unsigned short*)take((size_t)Mrows * NKVc * HDc * 2);
    unsigned short* vbb = (unsigned short*)take((size_t)Mrows * NKVc * HDc * 2);
    unsigned short* yb  = (unsigned short*)take((size_t)Mrows * NEc * 2);

    // 0) f32 -> bf16 converts
    cvt_f32_bf16<<<2048, 256, 0, stream>>>(x,  xb,  nX);
    cvt_f32_bf16<<<1024, 256, 0, stream>>>(Wq, wqb, nWq);
    cvt_f32_bf16<<<256,  256, 0, stream>>>(Wk, wkb, nWk);
    cvt_f32_bf16<<<256,  256, 0, stream>>>(Wv, wvb, nWk);
    cvt_f32_bf16<<<1024, 256, 0, stream>>>(Wo, wob, nWo);

    // 1) QKV projections (bf16 WMMA, f32 accumulate)
    gemm_bf16<<<dim3(NEc / 64, Mrows / 64), 256, 0, stream>>>(xb, wqb, qf, Mrows, NEc, NEc);
    gemm_bf16<<<dim3((NKVc * HDc) / 64, Mrows / 64), 256, 0, stream>>>(xb, wkb, kf, Mrows, NKVc * HDc, NEc);
    gemm_bf16<<<dim3((NKVc * HDc) / 64, Mrows / 64), 256, 0, stream>>>(xb, wvb, vf, Mrows, NKVc * HDc, NEc);

    // 2) rope + rmsnorm + gated value-embedding, emit bf16
    rope_norm_gate<<<Mrows, 128, 0, stream>>>(x, ve, cosb, sinb, Wg,
                                              qf, kf, vf, qbb, kbb, vbb);

    // 3) windowed-causal flash attention (WMMA for QK^T and PV)
    attn<<<dim3(Tc / 64, NHc, Bc), 128, 0, stream>>>(qbb, kbb, vbb, yb);

    // 4) output projection y @ Wo -> f32 out
    gemm_bf16<<<dim3(NEc / 64, Mrows / 64), 256, 0, stream>>>(yb, wob, out, Mrows, NEc, NEc);

    (void)in_sizes; (void)n_in; (void)out_size; (void)ws_size;
}
